// WindowAttention_60481729463047
// MI455X (gfx1250) — matrix operand order repaired and checked by reading, not compiled, not run on hardware
//
#include <hip/hip_runtime.h>

// ---------------------------------------------------------------------------
// Window attention for MI455X (gfx1250, wave32, WMMA f16 -> f32 accumulate).
// Pipeline: prep (f16 weight convert + bias tables) -> QKV WMMA GEMM ->
// per-(window,head) fused attention (WMMA QK^T, fp32 softmax, WMMA PV) ->
// proj WMMA GEMM.
// ---------------------------------------------------------------------------

typedef __attribute__((ext_vector_type(16))) _Float16 v16h;
typedef __attribute__((ext_vector_type(8)))  _Float16 v8h;
typedef __attribute__((ext_vector_type(4)))  _Float16 v4h;
typedef __attribute__((ext_vector_type(8)))  float    v8f;
typedef __attribute__((ext_vector_type(4)))  float    v4f;

#define NTOK   49
#define NHEAD  6
#define HD     32
#define DIM    192
#define QKVN   576           // 3*DIM
#define NWIN   2048
#define MROWS  (NWIN * NTOK) // 100352, divisible by 64
#define ASTR   200           // LDS row stride (halves) for 64x192 GEMM tiles
#define KSTR   40            // LDS row stride for K tile (64x32)
#define VSTR   72            // LDS row stride for V^T (32x64) and P (16x64)
#define SSTR   65            // LDS row stride (floats) for score rows

__device__ __forceinline__ v8f zero8f() {
  v8f z = {0.f,0.f,0.f,0.f,0.f,0.f,0.f,0.f};
  return z;
}
__device__ __forceinline__ v8h zero8h() {
  v8h z = {(_Float16)0.f,(_Float16)0.f,(_Float16)0.f,(_Float16)0.f,
           (_Float16)0.f,(_Float16)0.f,(_Float16)0.f,(_Float16)0.f};
  return z;
}
__device__ __forceinline__ v16h make16(v8h lo, v8h hi) {
  return __builtin_shufflevector(lo, hi, 0,1,2,3,4,5,6,7,8,9,10,11,12,13,14,15);
}
// A operand (16x32 f16): lane holds row (lane&15); halves [kb,kb+8) and
// [kb+16,kb+24) with kb = (lane>>4)*8.  `base` is row-major [m][k].
__device__ __forceinline__ v16h fragA(const _Float16* base, int stride, int k0) {
  int lane = (int)(threadIdx.x & 31u);
  const _Float16* p = base + (lane & 15) * stride + k0 + ((lane >> 4) << 3);
  return make16(*(const v8h*)p, *(const v8h*)(p + 16));
}
// B operand (32x16 f16): lane holds B^T row (lane&15); halves [kb,kb+8) and
// [kb+16,kb+24) with kb = k0 + (lane>>4)*8.  `base` is B^T row-major, i.e. [n][k].
__device__ __forceinline__ v16h fragB(const _Float16* base, int stride, int k0) {
  int lane = (int)(threadIdx.x & 31u);
  const _Float16* p = base + (lane & 15) * stride + k0 + ((lane >> 4) << 3);
  return make16(*(const v8h*)p, *(const v8h*)(p + 16));
}
__device__ __forceinline__ v8f wmma16(v16h a, v16h b, v8f c) {
  return __builtin_amdgcn_wmma_f32_16x16x32_f16(false, a, false, b,
                                                (short)0, c, false, false);
}

// ---------------------------------------------------------------------------
// prep: weight fp32->f16, A_phi[h][q][k] table, radial table Ar[b][h][r+6]
// ---------------------------------------------------------------------------
#define SEG_QKVW 110592      // 576*192
#define SEG_PRJW 36864       // 192*192
#define SEG_APHI 14406       // 6*49*49
#define SEG_ARAD 159744      // 2048*6*13
#define SEG_TOT  (SEG_QKVW + SEG_PRJW + SEG_APHI + SEG_ARAD)

__global__ void prep_kernel(const float* __restrict__ qkv_w,
                            const float* __restrict__ proj_w,
                            const float* __restrict__ a_p,
                            const float* __restrict__ b_p,
                            const float* __restrict__ a_r,
                            const float* __restrict__ b_r,
                            const float* __restrict__ D,
                            _Float16* __restrict__ qkvw_h,
                            _Float16* __restrict__ projw_h,
                            float* __restrict__ Aphi,
                            float* __restrict__ Artab) {
  int i = blockIdx.x * blockDim.x + threadIdx.x;
  const float PI = 3.14159265358979323846f;
  if (i >= SEG_TOT) return;
  if (i < SEG_QKVW) { qkvw_h[i] = (_Float16)qkv_w[i]; return; }
  i -= SEG_QKVW;
  if (i < SEG_PRJW) { projw_h[i] = (_Float16)proj_w[i]; return; }
  i -= SEG_PRJW;
  if (i < SEG_APHI) {
    int h = i / (NTOK * NTOK);
    int rem = i % (NTOK * NTOK);
    int q = rem / NTOK, k = rem % NTOK;
    int az = (q % 7) - (k % 7);                  // azimuth, [-6,6]
    int idx = az < 0 ? az + 13 : az;             // python-style mod 13
    float ang = (float)az * (2.f * PI / 56.f);
    Aphi[i] = a_p[idx * NHEAD + h] * cosf(ang) + b_p[idx * NHEAD + h] * sinf(ang);
    return;
  }
  i -= SEG_APHI;
  {
    int b = i / (NHEAD * 13);
    int rem = i % (NHEAD * 13);
    int h = rem / 13, r = rem % 13;
    int radius = r - 6;                          // [-6,6]
    int idx = radius < 0 ? radius + 13 : radius; // python-style mod 13
    float ang = D[b] * (float)radius * (2.f * PI / 224.f);
    Artab[i] = a_r[idx * NHEAD + h] * cosf(ang) + b_r[idx * NHEAD + h] * sinf(ang);
  }
}

// ---------------------------------------------------------------------------
// QKV GEMM: (100352 x 192 fp32) @ (192 x 576 f16) + bias -> f16, q scaled.
// 64x64 tile per 128-thread block (4 waves, one 16-row stripe per wave).
// ---------------------------------------------------------------------------
__global__ __launch_bounds__(128)
void qkv_gemm_kernel(const float* __restrict__ x,
                     const _Float16* __restrict__ wh,   // [576][192] f16
                     const float* __restrict__ qkv_b,
                     _Float16* __restrict__ qkvh) {     // [100352][576] f16
  __shared__ __align__(16) _Float16 At[64 * ASTR];
  __shared__ __align__(16) _Float16 Bt[64 * ASTR];
  const int tid = (int)threadIdx.x;
  const int gm0 = (int)blockIdx.x * 64;
  const int gn0 = (int)blockIdx.y * 64;

  // Stage A: 64x192 fp32 -> f16 (float4 loads, 24 per thread).
#pragma unroll
  for (int i = 0; i < 24; ++i) {
    int idx = i * 128 + tid;
    int row = idx / 48, c4 = (idx % 48) * 4;
    v4f f = *(const v4f*)(x + (size_t)(gm0 + row) * DIM + c4);
    v4h h4;
    h4[0] = (_Float16)f[0]; h4[1] = (_Float16)f[1];
    h4[2] = (_Float16)f[2]; h4[3] = (_Float16)f[3];
    *(v4h*)(At + row * ASTR + c4) = h4;
  }
  // Stage B^T rows (weight rows are already [outfeat][infeat]).
#pragma unroll
  for (int i = 0; i < 12; ++i) {
    int idx = i * 128 + tid;
    int row = idx / 24, c8 = (idx % 24) * 8;
    *(v8h*)(Bt + row * ASTR + c8) =
        *(const v8h*)(wh + (size_t)(gn0 + row) * DIM + c8);
  }
  __syncthreads();

  const int w = tid >> 5, lane = tid & 31;
  v8f acc[4];
#pragma unroll
  for (int i = 0; i < 4; ++i) acc[i] = zero8f();
  const _Float16* Arow = At + (w * 16) * ASTR;
#pragma unroll
  for (int kc = 0; kc < 6; ++kc) {
    int k0 = kc * 32;
    v16h a = fragA(Arow, ASTR, k0);
#pragma unroll
    for (int ns = 0; ns < 4; ++ns) {
      v16h b = fragB(Bt + ns * 16 * ASTR, ASTR, k0);
      acc[ns] = wmma16(a, b, acc[ns]);
    }
  }
  const float scale = 0.17677669529663687f;  // 32^-0.5
#pragma unroll
  for (int ns = 0; ns < 4; ++ns) {
    int gn = gn0 + ns * 16 + (lane & 15);
    float bv = qkv_b[gn];
    float mul = (gn < DIM) ? scale : 1.0f;   // q rows get the scale
#pragma unroll
    for (int v = 0; v < 8; ++v) {
      int gm = gm0 + w * 16 + v + ((lane >> 4) << 3);
      qkvh[(size_t)gm * QKVN + gn] = (_Float16)((acc[ns][v] + bv) * mul);
    }
  }
}

// ---------------------------------------------------------------------------
// Attention: one wave per (window b, head h).
// S = Q K^T (WMMA), + A_phi + A_r, fp32 softmax, out = P V (WMMA), f16 out.
// ---------------------------------------------------------------------------
__global__ __launch_bounds__(32)
void attn_kernel(const _Float16* __restrict__ qkvh,
                 const float* __restrict__ Aphi,
                 const float* __restrict__ Artab,
                 _Float16* __restrict__ attnh) {        // [100352][192] f16
  __shared__ __align__(16) _Float16 Kl[64 * KSTR];      // K, zero padded rows
  __shared__ __align__(16) _Float16 Vt[32 * VSTR];      // V^T [d][token]
  __shared__ __align__(16) _Float16 Pt[16 * VSTR];      // softmax probs, f16
  __shared__ float Sr[16 * SSTR];                       // scores, fp32
  __shared__ float Ar[13];
  const int b = (int)blockIdx.x, h = (int)blockIdx.y;
  const int lane = (int)threadIdx.x;
  const _Float16* qb = qkvh + (size_t)b * NTOK * QKVN + h * HD;
  const _Float16* kb = qb + DIM;
  const _Float16* vb = qb + 2 * DIM;

  // Stage K: 64 rows x 32 halves, zero-padded beyond 49 tokens.
#pragma unroll
  for (int i = 0; i < 8; ++i) {
    int idx = i * 32 + lane;          // 256 chunks of 8 halves
    int row = idx >> 2, c = (idx & 3) * 8;
    v8h val = zero8h();
    if (row < NTOK) val = *(const v8h*)(kb + (size_t)row * QKVN + c);
    *(v8h*)(Kl + row * KSTR + c) = val;
  }
  // Stage V^T: coalesced along d, zero-padded tokens.
  for (int t = 0; t < 64; ++t) {
    _Float16 val = (_Float16)0.f;
    if (t < NTOK) val = vb[(size_t)t * QKVN + lane];
    Vt[lane * VSTR + t] = val;
  }
  if (lane < 13) Ar[lane] = Artab[((size_t)b * NHEAD + h) * 13 + lane];
  __syncthreads();

  const float* AphiH = Aphi + h * NTOK * NTOK;
  for (int mt = 0; mt < 4; ++mt) {
    // Q fragment straight from global f16 (rows clamped to stay in window).
    int qr = mt * 16 + (lane & 15);
    if (qr > NTOK - 1) qr = NTOK - 1;
    int ks = (lane >> 4) << 3;
    const _Float16* qp = qb + (size_t)qr * QKVN + ks;
    v16h aq = make16(*(const v8h*)qp, *(const v8h*)(qp + 16));

#pragma unroll
    for (int nt = 0; nt < 4; ++nt) {
      v16h bk = fragB(Kl + nt * 16 * KSTR, KSTR, 0);
      v8f s = wmma16(aq, bk, zero8f());
      int n = nt * 16 + (lane & 15);
#pragma unroll
      for (int v = 0; v < 8; ++v) {
        int ml = v + ((lane >> 4) << 3);
        int mg = mt * 16 + ml;
        float val = -1e30f;
        if (mg < NTOK && n < NTOK) {
          int rad = mg / 7 - n / 7;  // query row - key row, [-6,6]
          val = s[v] + AphiH[mg * NTOK + n] + Ar[rad + 6];
        }
        Sr[ml * SSTR + n] = val;
      }
    }
    __syncthreads();

    // fp32 softmax, one row per lane (lanes 0..15); write f16 probs.
    if (lane < 16) {
      int mg = mt * 16 + lane;
      float* row = Sr + lane * SSTR;
      _Float16* prow = Pt + lane * VSTR;
      if (mg < NTOK) {
        float mx = -1e30f;
        for (int n = 0; n < NTOK; ++n) mx = fmaxf(mx, row[n]);
        float sum = 0.f;
        for (int n = 0; n < NTOK; ++n) {
          float e = __expf(row[n] - mx);
          row[n] = e;
          sum += e;
        }
        float inv = 1.f / sum;
        for (int n = 0; n < NTOK; ++n) prow[n] = (_Float16)(row[n] * inv);
        for (int n = NTOK; n < 64; ++n) prow[n] = (_Float16)0.f;
      } else {
        for (int n = 0; n < 64; ++n) prow[n] = (_Float16)0.f;
      }
    }
    __syncthreads();

    // out16x32 = P(16x64) @ V(64x32); chained WMMAs over padded K=64.
    v16h ap0 = fragA(Pt, VSTR, 0);
    v16h ap1 = fragA(Pt, VSTR, 32);
#pragma unroll
    for (int ns = 0; ns < 2; ++ns) {
      v16h b0 = fragB(Vt + ns * 16 * VSTR, VSTR, 0);
      v16h b1 = fragB(Vt + ns * 16 * VSTR, VSTR, 32);
      v8f o = wmma16(ap0, b0, zero8f());
      o = wmma16(ap1, b1, o);
      int col = h * HD + ns * 16 + (lane & 15);
#pragma unroll
      for (int v = 0; v < 8; ++v) {
        int mg = mt * 16 + v + ((lane >> 4) << 3);
        if (mg < NTOK)
          attnh[((size_t)b * NTOK + mg) * DIM + col] = (_Float16)o[v];
      }
    }
    __syncthreads();
  }
}

// ---------------------------------------------------------------------------
// Proj GEMM: (100352 x 192 f16) @ (192 x 192 f16) + bias -> fp32 output.
// ---------------------------------------------------------------------------
__global__ __launch_bounds__(128)
void proj_gemm_kernel(const _Float16* __restrict__ ah,   // [100352][192] f16
                      const _Float16* __restrict__ wh,   // [192][192] f16
                      const float* __restrict__ proj_b,
                      float* __restrict__ out) {         // [100352][192] fp32
  __shared__ __align__(16) _Float16 At[64 * ASTR];
  __shared__ __align__(16) _Float16 Bt[64 * ASTR];
  const int tid = (int)threadIdx.x;
  const int gm0 = (int)blockIdx.x * 64;
  const int gn0 = (int)blockIdx.y * 64;

#pragma unroll
  for (int i = 0; i < 12; ++i) {
    int idx = i * 128 + tid;
    int row = idx / 24, c8 = (idx % 24) * 8;
    *(v8h*)(At + row * ASTR + c8) =
        *(const v8h*)(ah + (size_t)(gm0 + row) * DIM + c8);
  }
#pragma unroll
  for (int i = 0; i < 12; ++i) {
    int idx = i * 128 + tid;
    int row = idx / 24, c8 = (idx % 24) * 8;
    *(v8h*)(Bt + row * ASTR + c8) =
        *(const v8h*)(wh + (size_t)(gn0 + row) * DIM + c8);
  }
  __syncthreads();

  const int w = tid >> 5, lane = tid & 31;
  v8f acc[4];
#pragma unroll
  for (int i = 0; i < 4; ++i) acc[i] = zero8f();
  const _Float16* Arow = At + (w * 16) * ASTR;
#pragma unroll
  for (int kc = 0; kc < 6; ++kc) {
    int k0 = kc * 32;
    v16h a = fragA(Arow, ASTR, k0);
#pragma unroll
    for (int ns = 0; ns < 4; ++ns) {
      v16h b = fragB(Bt + ns * 16 * ASTR, ASTR, k0);
      acc[ns] = wmma16(a, b, acc[ns]);
    }
  }
#pragma unroll
  for (int ns = 0; ns < 4; ++ns) {
    int gn = gn0 + ns * 16 + (lane & 15);
    float bv = proj_b[gn];
#pragma unroll
    for (int v = 0; v < 8; ++v) {
      int gm = gm0 + w * 16 + v + ((lane >> 4) << 3);
      out[(size_t)gm * DIM + gn] = acc[ns][v] + bv;
    }
  }
}

// ---------------------------------------------------------------------------
// Host launcher
// ---------------------------------------------------------------------------
extern "C" void kernel_launch(void* const* d_in, const int* in_sizes, int n_in,
                              void* d_out, int out_size, void* d_ws, size_t ws_size,
                              hipStream_t stream) {
  (void)in_sizes; (void)n_in; (void)out_size; (void)ws_size;
  const float* x      = (const float*)d_in[0];
  const float* D      = (const float*)d_in[1];
  const float* qkv_w  = (const float*)d_in[2];
  const float* qkv_b  = (const float*)d_in[3];
  const float* proj_w = (const float*)d_in[4];
  const float* proj_b = (const float*)d_in[5];
  const float* a_p    = (const float*)d_in[6];
  const float* b_p    = (const float*)d_in[7];
  const float* a_r    = (const float*)d_in[8];
  const float* b_r    = (const float*)d_in[9];
  float* out = (float*)d_out;

  char* ws = (char*)d_ws;
  size_t off = 0;
  auto take = [&](size_t bytes) -> char* {
    char* p = ws + off;
    off = (off + bytes + 255) & ~(size_t)255;
    return p;
  };
  _Float16* qkvw_h  = (_Float16*)take((size_t)SEG_QKVW * 2);
  _Float16* projw_h = (_Float16*)take((size_t)SEG_PRJW * 2);
  float*    Aphi    = (float*)take((size_t)SEG_APHI * 4);
  float*    Artab   = (float*)take((size_t)SEG_ARAD * 4);
  _Float16* qkvh    = (_Float16*)take((size_t)MROWS * QKVN * 2);
  _Float16* attnh   = (_Float16*)take((size_t)MROWS * DIM * 2);

  prep_kernel<<<(SEG_TOT + 255) / 256, 256, 0, stream>>>(
      qkv_w, proj_w, a_p, b_p, a_r, b_r, D, qkvw_h, projw_h, Aphi, Artab);
  qkv_gemm_kernel<<<dim3(MROWS / 64, QKVN / 64), 128, 0, stream>>>(
      x, qkvw_h, qkv_b, qkvh);
  attn_kernel<<<dim3(NWIN, NHEAD), 32, 0, stream>>>(qkvh, Aphi, Artab, attnh);
  proj_gemm_kernel<<<dim3(MROWS / 64, DIM / 64), 128, 0, stream>>>(
      attnh, projw_h, proj_b, out);
}
